// Seq2Seq2_65360812310785
// MI455X (gfx1250) — matrix operand order repaired and checked by reading, not compiled, not run on hardware
//
#include <hip/hip_runtime.h>
#include <hip/hip_bf16.h>

// Problem constants (from reference)
#define B_SZ   128
#define T_IN   256
#define D_IN   64
#define H_SZ   1024
#define FOURH  4096
#define T_OUT  128
#define D_OUT  128

typedef __bf16 bf16;
typedef __attribute__((ext_vector_type(16))) __bf16 v16bf;
typedef __attribute__((ext_vector_type(8)))  float   v8f;

union ABFrag { v16bf v; uint4 q[2]; };

// A fragment: 16x32 (MxK) bf16, A row-major [M, lda] in GLOBAL memory.
// ISA layout: lane<16 -> row M=lane, K = k0+0..7 (V0-3) and k0+16..23 (V4-7)
//             lane>=16 -> row M=lane-16, K = k0+8..15 and k0+24..31
__device__ __forceinline__ v16bf load_a_frag(const bf16* __restrict__ A, int lda,
                                             int mBase, int k0, int lane) {
  const int hi = lane >> 4;
  const int m  = mBase + (lane & 15);
  const bf16* p = A + (size_t)m * lda + k0 + hi * 8;
  ABFrag r;
  r.q[0] = *(const uint4*)(p);
  r.q[1] = *(const uint4*)(p + 16);
  return r.v;
}

// A fragment sourced from the LDS-staged tile (K-chunk already at columns 0..31).
// Row stride 40 bf16 = 80B (multiple of 16B, and 20-dword rotation -> no bank conflicts).
__device__ __forceinline__ v16bf lds_a_frag(const bf16 (*tile)[40], int mOff, int lane) {
  const int hi = lane >> 4;
  const int m  = mOff + (lane & 15);
  ABFrag r;
  r.q[0] = *(const uint4*)&tile[m][hi * 8];
  r.q[1] = *(const uint4*)&tile[m][hi * 8 + 16];
  return r.v;
}

// B fragment: 32x16 (KxN) bf16, from pre-transposed BT [N, ldb] (row-major in K).
// ISA layout: lane<16 -> col N=lane, K=k0+0..7 and k0+16..23 ; lane>=16 -> col N=lane-16, K=k0+8..15 and k0+24..31
__device__ __forceinline__ v16bf load_b_frag(const bf16* __restrict__ BT, int ldb,
                                             int nBase, int k0, int lane) {
  const int hi = lane >> 4;
  const int n  = nBase + (lane & 15);
  const bf16* p = BT + (size_t)n * ldb + k0 + hi * 8;
  ABFrag r;
  r.q[0] = *(const uint4*)(p);
  r.q[1] = *(const uint4*)(p + 16);
  return r.v;
}

// CDNA5 async global->LDS copy (ASYNCcnt path), portable via inline asm.
__device__ __forceinline__ void async_load_b128_to_lds(unsigned lds_addr, const void* gptr) {
  asm volatile("global_load_async_to_lds_b128 %0, %1, off"
               :: "v"(lds_addr), "v"((unsigned long long)(size_t)gptr)
               : "memory");
}
__device__ __forceinline__ void wait_asynccnt0() {
  asm volatile("s_wait_asynccnt 0x0" ::: "memory");
}

__device__ __forceinline__ float fsigmoid(float x) { return 1.0f / (1.0f + __expf(-x)); }

// ---------------------------------------------------------------------------
// One LSTM timestep, fully fused. Grid: (H/16, B/64) x 256 threads (8 waves).
// Block covers 16 H-columns (x all 4 gates) and 64 batch rows.
// Wave w: gate = w&3, mHalf = w>>2. Each wave owns TWO 16x16 M-tiles sharing
// one B fragment per k-iteration (2 WMMAs per B load). The A (h) tile for the
// whole block is staged into a DOUBLE-BUFFERED LDS tile with async-to-LDS
// DMA: chunk k+1 streams in while chunk k is consumed by the tensor cores,
// so there is one barrier per k-iteration and the DMA is off the critical path.
// If gatesOut != null: store fp32 pre-activations and skip the cell update.
// ---------------------------------------------------------------------------
__global__ __launch_bounds__(256) void lstm_step_wmma(
    const bf16* __restrict__ hA,      // [B, H] bf16
    const bf16* __restrict__ WhT,     // [4H, H] bf16 (transposed weights)
    const bf16* __restrict__ xA,      // [B, K2] bf16 with row stride ldx (nullable)
    const bf16* __restrict__ WiT,     // [4H, K2] bf16
    int K2, int ldx,
    const float* __restrict__ pre,    // bias [4H] (preIsMatrix=0) or matrix [B,4H]
    int preIsMatrix,
    float* __restrict__ cSt,          // [B, H] fp32 cell state (in/out)
    bf16* __restrict__ hOut,          // [B, H] bf16 (must differ from hA)
    bf16* __restrict__ hsOut,         // optional trajectory store (pre-offset), stride ldhs
    int ldhs,
    float* __restrict__ gatesOut)     // optional fp32 [B,4H] store mode
{
  __shared__ bf16  tileA[2][64][40];   // double-buffered: 2 x (64 rows x 32 K, pad 40) = 10 KB
  __shared__ float lds_g[4][64][16];   // gate exchange, 16 KB

  const int n0   = blockIdx.x * 16;    // column slice within H
  const int m0   = blockIdx.y * 64;    // batch row slice (64 rows)
  const int tid  = threadIdx.x;
  const int wave = tid >> 5;
  const int lane = tid & 31;
  const int gate = wave & 3;
  const int mW   = (wave >> 2) * 32;   // wave's row offset inside block (0 or 32)
  const int ng   = gate * H_SZ + n0;   // column in the 4H gate space

  const int hi   = lane >> 4;
  const int ccol = lane & 15;
  const int crow = hi * 8;             // C layout: VGPR r -> row r + 8*hi

  // Init accumulators with the additive term (bias or precomputed xg matrix)
  v8f acc0, acc1;
#pragma unroll
  for (int r = 0; r < 8; ++r) {
    if (preIsMatrix) {
      acc0[r] = pre[(size_t)(m0 + mW + crow + r) * FOURH + ng + ccol];
      acc1[r] = pre[(size_t)(m0 + mW + 16 + crow + r) * FOURH + ng + ccol];
    } else {
      acc0[r] = pre[ng + ccol];
      acc1[r] = acc0[r];
    }
  }

  // Per-thread staging slot: 64 rows x 64B -> 256 chunks of 16B, one per thread
  const int sRow  = tid >> 2;          // 0..63
  const int sPart = (tid & 3) * 8;     // bf16 element offset 0/8/16/24
  const unsigned ldsAddr0 = (unsigned)(size_t)&tileA[0][sRow][sPart];
  const unsigned ldsAddr1 = (unsigned)(size_t)&tileA[1][sRow][sPart];
  const bf16* gRow = hA + (size_t)(m0 + sRow) * H_SZ + sPart;

  // Prologue: stage first K-chunk into buffer 0
  async_load_b128_to_lds(ldsAddr0, gRow);

  // Recurrent GEMM: K = H, 32 per WMMA, A double-buffered through LDS
  for (int k0 = 0; k0 < H_SZ; k0 += 32) {
    const int cur = (k0 >> 5) & 1;
    wait_asynccnt0();                  // my chunk for buf[cur] has landed
    __syncthreads();                   // everyone's chunk landed; buf[nxt] readers done
    if (k0 + 32 < H_SZ)                // overlap next DMA with this chunk's WMMAs
      async_load_b128_to_lds(cur ? ldsAddr0 : ldsAddr1, gRow + k0 + 32);
    if (k0 + 64 < H_SZ)
      __builtin_prefetch(WhT + (size_t)(ng + ccol) * H_SZ + k0 + 64, 0, 1);
    v16bf b  = load_b_frag(WhT, H_SZ, ng, k0, lane);
    v16bf a0 = lds_a_frag(tileA[cur], mW, lane);
    v16bf a1 = lds_a_frag(tileA[cur], mW + 16, lane);
    acc0 = __builtin_amdgcn_wmma_f32_16x16x32_bf16(false, a0, false, b,
                                                   (short)0, acc0, false, false);
    acc1 = __builtin_amdgcn_wmma_f32_16x16x32_bf16(false, a1, false, b,
                                                   (short)0, acc1, false, false);
  }

  // Optional fused input projection (layer 1: x_t @ Wi1, K2 = 64), direct global
  if (xA) {
#pragma unroll
    for (int k0 = 0; k0 < 64; k0 += 32) {
      if (k0 >= K2) break;
      v16bf b  = load_b_frag(WiT, K2, ng, k0, lane);
      v16bf a0 = load_a_frag(xA, ldx, m0 + mW, k0, lane);
      v16bf a1 = load_a_frag(xA, ldx, m0 + mW + 16, k0, lane);
      acc0 = __builtin_amdgcn_wmma_f32_16x16x32_bf16(false, a0, false, b,
                                                     (short)0, acc0, false, false);
      acc1 = __builtin_amdgcn_wmma_f32_16x16x32_bf16(false, a1, false, b,
                                                     (short)0, acc1, false, false);
    }
  }

  if (gatesOut) {  // pre-activation store mode (xg2 precompute)
#pragma unroll
    for (int r = 0; r < 8; ++r) {
      gatesOut[(size_t)(m0 + mW + crow + r) * FOURH + ng + ccol]      = acc0[r];
      gatesOut[(size_t)(m0 + mW + 16 + crow + r) * FOURH + ng + ccol] = acc1[r];
    }
    return;
  }

  // Exchange gate tiles through LDS so each thread sees i,f,g,o for its element
#pragma unroll
  for (int r = 0; r < 8; ++r) {
    lds_g[gate][mW + crow + r][ccol]      = acc0[r];
    lds_g[gate][mW + 16 + crow + r][ccol] = acc1[r];
  }
  __syncthreads();

  // 64 rows x 16 cols = 1024 elements, 256 threads -> 4 each
  for (int e = tid; e < 1024; e += 256) {
    const int r  = e >> 4;
    const int cc = e & 15;
    const float gi = lds_g[0][r][cc];
    const float gf = lds_g[1][r][cc];
    const float gg = lds_g[2][r][cc];
    const float go = lds_g[3][r][cc];
    const int    bb  = m0 + r;
    const int    col = n0 + cc;
    const size_t idx = (size_t)bb * H_SZ + col;
    const float cOld = cSt[idx];
    const float cNew = fsigmoid(gf) * cOld + fsigmoid(gi) * tanhf(gg);
    const float hNew = fsigmoid(go) * tanhf(cNew);
    cSt[idx]  = cNew;
    const bf16 hb = (bf16)hNew;
    hOut[idx] = hb;
    if (hsOut) hsOut[(size_t)bb * ldhs + col] = hb;
  }
}

// ---------------------------------------------------------------------------
// Final linear: out[M=16384, 128] = hs2[M, 1024] @ WlinT'[128,1024] + blin
// Grid: (M/128, 128/16) x 256 threads; wave w owns m-tile w of the 128-row block
// ---------------------------------------------------------------------------
__global__ __launch_bounds__(256) void linout_wmma(
    const bf16* __restrict__ A,    // [M, 1024]
    const bf16* __restrict__ BT,   // [128, 1024]
    const float* __restrict__ bias,
    float* __restrict__ out)       // [M, 128]
{
  const int wave = threadIdx.x >> 5;
  const int lane = threadIdx.x & 31;
  const int mBase = blockIdx.x * 128 + wave * 16;
  const int n0    = blockIdx.y * 16;
  const int hi   = lane >> 4;
  const int ccol = lane & 15;
  const float bv = bias[n0 + ccol];

  v8f acc;
#pragma unroll
  for (int r = 0; r < 8; ++r) acc[r] = bv;

#pragma unroll 4
  for (int k0 = 0; k0 < H_SZ; k0 += 32) {
    v16bf a = load_a_frag(A, H_SZ, mBase, k0, lane);
    v16bf b = load_b_frag(BT, H_SZ, n0, k0, lane);
    acc = __builtin_amdgcn_wmma_f32_16x16x32_bf16(false, a, false, b,
                                                  (short)0, acc, false, false);
  }
#pragma unroll
  for (int r = 0; r < 8; ++r)
    out[(size_t)(mBase + hi * 8 + r) * D_OUT + n0 + ccol] = acc[r];
}

// ---------------------------------------------------------------------------
// Prep kernels (run once per launch; deterministic rebuild of workspace)
// ---------------------------------------------------------------------------
__global__ void transpose_to_bf16(const float* __restrict__ src, bf16* __restrict__ dst,
                                  int R, int C) {
  // src [R][C] row-major -> dst [C][R] bf16
  for (size_t i = (size_t)blockIdx.x * blockDim.x + threadIdx.x;
       i < (size_t)R * C; i += (size_t)gridDim.x * blockDim.x) {
    const int rr = (int)(i / C);
    const int cc = (int)(i % C);
    dst[(size_t)cc * R + rr] = (bf16)src[i];
  }
}

__global__ void convert_to_bf16(const float* __restrict__ src, bf16* __restrict__ dst,
                                size_t n) {
  for (size_t i = (size_t)blockIdx.x * blockDim.x + threadIdx.x;
       i < n; i += (size_t)gridDim.x * blockDim.x)
    dst[i] = (bf16)src[i];
}

__global__ void zero_state(bf16* __restrict__ h, float* __restrict__ c, size_t n) {
  for (size_t i = (size_t)blockIdx.x * blockDim.x + threadIdx.x;
       i < n; i += (size_t)gridDim.x * blockDim.x) {
    h[i] = (bf16)0.0f;
    c[i] = 0.0f;
  }
}

// ---------------------------------------------------------------------------
extern "C" void kernel_launch(void* const* d_in, const int* in_sizes, int n_in,
                              void* d_out, int out_size, void* d_ws, size_t ws_size,
                              hipStream_t stream) {
  const float* x    = (const float*)d_in[0];  // [B, TIN, DIN]
  const float* Wi1  = (const float*)d_in[1];  // [DIN, 4H]
  const float* Wh1  = (const float*)d_in[2];  // [H, 4H]
  const float* b1   = (const float*)d_in[3];  // [4H]
  const float* Wi2  = (const float*)d_in[4];  // [H, 4H]
  const float* Wh2  = (const float*)d_in[5];  // [H, 4H]
  const float* b2   = (const float*)d_in[6];  // [4H]
  const float* Wlin = (const float*)d_in[7];  // [H, DOUT]
  const float* blin = (const float*)d_in[8];  // [DOUT]
  float* out = (float*)d_out;                 // [B, TOUT, DOUT]

  // Workspace carve (256B aligned slices)
  char* w = (char*)d_ws;
  size_t off = 0;
  auto carve = [&](size_t bytes) -> char* {
    char* p = w + off;
    off += (bytes + 255) & ~(size_t)255;
    return p;
  };
  bf16*  WiT1  = (bf16*)carve((size_t)FOURH * D_IN * 2);       // [4H, 64]
  bf16*  WhT1  = (bf16*)carve((size_t)FOURH * H_SZ * 2);       // [4H, H]
  bf16*  WiT2  = (bf16*)carve((size_t)FOURH * H_SZ * 2);       // [4H, H]
  bf16*  WhT2  = (bf16*)carve((size_t)FOURH * H_SZ * 2);       // [4H, H]
  bf16*  WlinT = (bf16*)carve((size_t)D_OUT * H_SZ * 2);       // [128, H]
  bf16*  xbf   = (bf16*)carve((size_t)B_SZ * T_IN * D_IN * 2); // [B, TIN, DIN]
  bf16*  hbuf0 = (bf16*)carve((size_t)B_SZ * H_SZ * 2);
  bf16*  hbuf1 = (bf16*)carve((size_t)B_SZ * H_SZ * 2);
  float* cbuf  = (float*)carve((size_t)B_SZ * H_SZ * 4);
  float* xg2   = (float*)carve((size_t)B_SZ * FOURH * 4);
  bf16*  hs2   = (bf16*)carve((size_t)B_SZ * T_OUT * H_SZ * 2);

  // --- weight / input conversion (deterministic per call) ---
  transpose_to_bf16<<<512, 256, 0, stream>>>(Wi1,  WiT1,  D_IN, FOURH);
  transpose_to_bf16<<<1024, 256, 0, stream>>>(Wh1, WhT1,  H_SZ, FOURH);
  transpose_to_bf16<<<1024, 256, 0, stream>>>(Wi2, WiT2,  H_SZ, FOURH);
  transpose_to_bf16<<<1024, 256, 0, stream>>>(Wh2, WhT2,  H_SZ, FOURH);
  transpose_to_bf16<<<256, 256, 0, stream>>>(Wlin, WlinT, H_SZ, D_OUT);
  convert_to_bf16<<<1024, 256, 0, stream>>>(x, xbf, (size_t)B_SZ * T_IN * D_IN);
  zero_state<<<256, 256, 0, stream>>>(hbuf0, cbuf, (size_t)B_SZ * H_SZ);

  const dim3 stepGrid(H_SZ / 16, B_SZ / 64);  // (64, 2)

  // --- layer 1: 256 sequential steps, x_t@Wi1 fused as extra K=64 ---
  bf16* hIn = hbuf0;
  bf16* hOt = hbuf1;
  for (int t = 0; t < T_IN; ++t) {
    lstm_step_wmma<<<stepGrid, 256, 0, stream>>>(
        hIn, WhT1, xbf + (size_t)t * D_IN, WiT1, D_IN, T_IN * D_IN,
        b1, 0, cbuf, hOt, nullptr, 0, nullptr);
    bf16* tmp = hIn; hIn = hOt; hOt = tmp;
  }
  // hIn now holds h_T == t_last ; cbuf holds c_T

  // --- xg2 = t_last @ Wi2 + b2, computed ONCE (z is a constant broadcast) ---
  lstm_step_wmma<<<stepGrid, 256, 0, stream>>>(
      hIn, WiT2, nullptr, nullptr, 0, 0,
      b2, 0, nullptr, nullptr, nullptr, 0, xg2);

  // --- layer 2: 128 sequential steps, pre = xg2 matrix; record hs2 ---
  for (int t = 0; t < T_OUT; ++t) {
    lstm_step_wmma<<<stepGrid, 256, 0, stream>>>(
        hIn, WhT2, nullptr, nullptr, 0, 0,
        xg2, 1, cbuf, hOt, hs2 + (size_t)t * H_SZ, T_OUT * H_SZ, nullptr);
    bf16* tmp = hIn; hIn = hOt; hOt = tmp;
  }

  // --- final linear: [B*TOUT, H] @ [H, DOUT] + blin ---
  linout_wmma<<<dim3((B_SZ * T_OUT) / 128, D_OUT / 16), 256, 0, stream>>>(
      hs2, WlinT, blin, out);
}
